// BasicTransformerBlock_14190571946001
// MI455X (gfx1250) — matrix operand order repaired and checked by reading, not compiled, not run on hardware
//
#include <hip/hip_runtime.h>
#include <hip/hip_bf16.h>
#include <stdint.h>

// ---------------- problem constants ----------------
#define DIMC   1280
#define HEADS  8
#define HDIM   160
#define CTXD   768
#define NB     4
#define SEQ    1024
#define TCTX   77
#define TPADC  96          // padded cross-attn T
#define ROWS   (NB*SEQ)    // 4096
#define CROWS  (NB*TCTX)   // 308
#define FF     (4*DIMC)    // 5120
#define FF2    (8*DIMC)    // 10240

typedef __attribute__((ext_vector_type(16))) __bf16 v16bf;
typedef __attribute__((ext_vector_type(8)))  float  v8f;
typedef unsigned short u16;

__device__ __forceinline__ u16 f2bf(float f) {
  union { float f; unsigned int u; } c; c.f = f;
  unsigned int u = c.u;
  u += 0x7FFFu + ((u >> 16) & 1u);   // round-to-nearest-even
  return (u16)(u >> 16);
}
__device__ __forceinline__ float bf2f(u16 h) {
  union { unsigned int u; float f; } c; c.u = ((unsigned int)h) << 16;
  return c.f;
}

// CDNA5 async global->LDS copy (ASYNCcnt-tracked, no VGPR staging).
// lds operand is the as3 byte offset (= low 32
// bits of the generic pointer to a __shared__ object).
__device__ __forceinline__ void async_b128(unsigned ldsoff, const u16* gp) {
  asm volatile("global_load_async_to_lds_b128 %0, %1, off"
               :: "v"(ldsoff), "v"(gp) : "memory");
}
// Wait until ASYNCcnt <= n (n is 3 while one pipelined tile is in flight).
__device__ __forceinline__ void s_wait_async(int pend) {
  if (pend) asm volatile("s_wait_asynccnt 3" ::: "memory");
  else      asm volatile("s_wait_asynccnt 0" ::: "memory");
}

// -------- weight transpose-convert: W[K,N] f32 -> Wt[N,K] bf16 --------
__global__ void k_wtrans(const float* __restrict__ src, u16* __restrict__ dst,
                         int K, int N) {
  long long idx = (long long)blockIdx.x * blockDim.x + threadIdx.x;
  long long total = (long long)K * N;
  if (idx >= total) return;
  int n = (int)(idx / K);
  int k = (int)(idx - (long long)n * K);
  dst[idx] = f2bf(src[(long long)k * N + n]);
}

__global__ void k_tobf16(const float* __restrict__ src, u16* __restrict__ dst, int n) {
  int i = blockIdx.x * blockDim.x + threadIdx.x;
  if (i < n) dst[i] = f2bf(src[i]);
}

// -------- LayerNorm: one block (256 thr) per row of D=1280, bf16 out --------
__global__ void __launch_bounds__(256) k_layernorm(
    const float* __restrict__ x, const float* __restrict__ g,
    const float* __restrict__ b, u16* __restrict__ out, int D) {
  __shared__ float r1[256], r2[256];
  const int row = blockIdx.x, tid = threadIdx.x;
  const float* xr = x + (long long)row * D;
  float v[5]; float s = 0.f, s2 = 0.f;
  #pragma unroll
  for (int j = 0; j < 5; ++j) {
    int i = tid + j * 256;
    v[j] = (i < D) ? xr[i] : 0.f;
    s += v[j]; s2 += v[j] * v[j];
  }
  r1[tid] = s; r2[tid] = s2; __syncthreads();
  for (int st = 128; st > 0; st >>= 1) {
    if (tid < st) { r1[tid] += r1[tid + st]; r2[tid] += r2[tid + st]; }
    __syncthreads();
  }
  float mu = r1[0] / D;
  float var = r2[0] / D - mu * mu;
  float rs = rsqrtf(var + 1e-5f);
  #pragma unroll
  for (int j = 0; j < 5; ++j) {
    int i = tid + j * 256;
    if (i < D) out[(long long)row * D + i] = f2bf((v[j] - mu) * rs * g[i] + b[i]);
  }
}

// -------- row softmax: scores f32 [row][ld] -> P bf16 [row][ld] --------
__global__ void __launch_bounds__(256) k_softmax(
    const float* __restrict__ S, u16* __restrict__ P, int T, int ld, float scale) {
  __shared__ float red[256];
  const int row = blockIdx.x, tid = threadIdx.x;
  const float* sr = S + (long long)row * ld;
  float v[4]; float mx = -3.0e38f;
  #pragma unroll
  for (int j = 0; j < 4; ++j) {
    int i = tid + j * 256;
    v[j] = (i < T) ? sr[i] * scale : -3.0e38f;
    mx = fmaxf(mx, v[j]);
  }
  red[tid] = mx; __syncthreads();
  for (int st = 128; st > 0; st >>= 1) {
    if (tid < st) red[tid] = fmaxf(red[tid], red[tid + st]);
    __syncthreads();
  }
  mx = red[0]; __syncthreads();
  float sum = 0.f;
  #pragma unroll
  for (int j = 0; j < 4; ++j) {
    int i = tid + j * 256;
    float e = (i < T) ? expf(v[j] - mx) : 0.f;
    v[j] = e; sum += e;
  }
  red[tid] = sum; __syncthreads();
  for (int st = 128; st > 0; st >>= 1) {
    if (tid < st) red[tid] += red[tid + st];
    __syncthreads();
  }
  float inv = 1.f / red[0];
  #pragma unroll
  for (int j = 0; j < 4; ++j) {
    int i = tid + j * 256;
    if (i < T) P[(long long)row * ld + i] = f2bf(v[j] * inv);
  }
}

// -------- GEGLU gate: h bf16 [M][8D] -> out bf16 [M][4D] --------
__global__ void k_geglu(const u16* __restrict__ h, u16* __restrict__ out, int n) {
  int idx = blockIdx.x * blockDim.x + threadIdx.x;
  if (idx >= n) return;
  int m = idx / FF, c = idx - m * FF;
  float val  = bf2f(h[(long long)m * FF2 + c]);
  float gate = bf2f(h[(long long)m * FF2 + FF + c]);
  float t = tanhf(gate * 0.7978845608f * (1.f + 0.044715f * gate * gate));
  out[idx] = f2bf(val * 0.5f * gate * (1.f + t));
}

// ================= WMMA GEMM: C = A[M,K] * Bt[N,K]^T (+bias)(+res) =================
// OM: 0 = f32 out, 1 = bf16 out, 2 = bf16 transposed out (C[n][colmap(m)])
// Double-buffered LDS; interior tiles staged with global_load_async_to_lds_b128,
// edge tiles (M/N/K tails) via guarded register staging. Uniform branch selects.
#define BM 128
#define BN 64
#define BK 32
#define LDSK 40   // padded LDS K stride (80B rows -> 16B aligned, conflict-free frags)

template<int OM, bool HB, bool HR>
__global__ void __launch_bounds__(256) k_gemm(
    const u16* __restrict__ A, int lda,
    const u16* __restrict__ Bt, int ldb,
    void* __restrict__ Cp, int ldc,
    const float* __restrict__ bias,
    const float* __restrict__ Res, int ldr,
    int M, int N, int K, int batchH,
    long long aSb, long long aSh, long long bSb, long long bSh,
    long long cSb, long long cSh, int mSeg, int mPad)
{
  __shared__ u16 sA[2][BM * LDSK];
  __shared__ u16 sB[2][BN * LDSK];

  const int z  = blockIdx.z;
  const int bb = z / batchH;
  const int hh = z - bb * batchH;
  A  += bb * aSb + hh * aSh;
  Bt += bb * bSb + hh * bSh;
  const long long cOff = bb * cSb + hh * cSh;

  const int tid  = threadIdx.x;
  const int wave = tid >> 5;
  const int lane = tid & 31;
  const int wm = wave & 3;   // 4 waves along M
  const int wn = wave >> 2;  // 2 waves along N
  const int m0 = blockIdx.y * BM;
  const int n0 = blockIdx.x * BN;

  v8f acc[2][2];
  #pragma unroll
  for (int a = 0; a < 2; ++a)
    #pragma unroll
    for (int b = 0; b < 2; ++b)
      #pragma unroll
      for (int e = 0; e < 8; ++e) acc[a][b][e] = 0.f;

  const int nT = (K + BK - 1) / BK;

  auto loadTile = [&](int t, int buf) -> int {
    const int k0 = t * BK;
    u16* dA = sA[buf];
    u16* dB = sB[buf];
    if ((m0 + BM <= M) && (n0 + BN <= N) && (k0 + BK <= K)) {
      // fully interior: async LDS-direct copy, 3 x b128 per thread
      #pragma unroll
      for (int i = 0; i < 2; ++i) {
        const int c = tid + i * 256;
        const int r = c >> 2, cc = (c & 3) << 3;
        async_b128((unsigned)(unsigned long long)(const void*)&dA[r * LDSK + cc],
                   A + (long long)(m0 + r) * lda + (k0 + cc));
      }
      {
        const int r = tid >> 2, cc = (tid & 3) << 3;
        async_b128((unsigned)(unsigned long long)(const void*)&dB[r * LDSK + cc],
                   Bt + (long long)(n0 + r) * ldb + (k0 + cc));
      }
      return 3;
    }
    // guarded fallback for M/N/K tails
    #pragma unroll
    for (int i = 0; i < 2; ++i) {
      int c = tid + i * 256;
      int r = c >> 2, cc = (c & 3) << 3;
      int gm = m0 + r, gk = k0 + cc;
      const u16* ap = A + (long long)gm * lda + gk;
      u16 tmp[8];
      if (gm < M && (gk + 7) < K) {
        *(int4*)tmp = *(const int4*)ap;
      } else {
        #pragma unroll
        for (int j = 0; j < 8; ++j)
          tmp[j] = (gm < M && (gk + j) < K) ? ap[j] : (u16)0;
      }
      *(int4*)&dA[r * LDSK + cc] = *(const int4*)tmp;
    }
    {
      int r = tid >> 2, cc = (tid & 3) << 3;
      int gn = n0 + r, gk = k0 + cc;
      const u16* bp = Bt + (long long)gn * ldb + gk;
      u16 tmp[8];
      if (gn < N && (gk + 7) < K) {
        *(int4*)tmp = *(const int4*)bp;
      } else {
        #pragma unroll
        for (int j = 0; j < 8; ++j)
          tmp[j] = (gn < N && (gk + j) < K) ? bp[j] : (u16)0;
      }
      *(int4*)&dB[r * LDSK + cc] = *(const int4*)tmp;
    }
    return 0;
  };

  (void)loadTile(0, 0);   // prologue

  const int mrow = lane & 15;
  const int hi   = lane >> 4;

  for (int t = 0; t < nT; ++t) {
    const int buf = t & 1;
    int nextPend = 0;
    if (t + 1 < nT) nextPend = loadTile(t + 1, buf ^ 1);
    s_wait_async(nextPend);   // my tile-t async copies complete (in-order)
    __syncthreads();          // tile-t visible across all waves

    union { v16bf v; unsigned int u[8]; } af[2], bfr[2];
    #pragma unroll
    for (int tm = 0; tm < 2; ++tm) {
      const u16* base = &sA[buf][(wm * 32 + tm * 16 + mrow) * LDSK];
      #pragma unroll
      for (int r = 0; r < 8; ++r) {
        int kk = (hi ? 8 : 0) + ((r < 4) ? (2 * r) : (16 + 2 * (r - 4)));
        af[tm].u[r] = *(const unsigned int*)(base + kk);
      }
    }
    #pragma unroll
    for (int tn = 0; tn < 2; ++tn) {
      const u16* base = &sB[buf][(wn * 32 + tn * 16 + mrow) * LDSK];
      #pragma unroll
      for (int r = 0; r < 8; ++r) {
        int kk = (hi ? 8 : 0) + ((r < 4) ? (2 * r) : (16 + 2 * (r - 4)));
        bfr[tn].u[r] = *(const unsigned int*)(base + kk);
      }
    }
    #pragma unroll
    for (int tm = 0; tm < 2; ++tm)
      #pragma unroll
      for (int tn = 0; tn < 2; ++tn)
        acc[tm][tn] = __builtin_amdgcn_wmma_f32_16x16x32_bf16(
            false, af[tm].v, false, bfr[tn].v, (short)0, acc[tm][tn], false, false);

    __syncthreads();          // all waves done reading buf before it is refilled
  }

  // ---- epilogue: C/D layout m = r + 8*(lane/16), n = lane%16 ----
  const int nl = lane & 15;
  const int mo = (lane >> 4) * 8;
  #pragma unroll
  for (int tm = 0; tm < 2; ++tm)
    #pragma unroll
    for (int tn = 0; tn < 2; ++tn)
      #pragma unroll
      for (int r = 0; r < 8; ++r) {
        int gm = m0 + wm * 32 + tm * 16 + mo + r;
        int gn = n0 + wn * 32 + tn * 16 + nl;
        if (gm >= M || gn >= N) continue;
        float v = acc[tm][tn][r];
        if (HB) v += bias[gn];
        if (HR) v += Res[(long long)gm * ldr + gn];
        if (OM == 0) {
          ((float*)Cp)[cOff + (long long)gm * ldc + gn] = v;
        } else if (OM == 1) {
          ((u16*)Cp)[cOff + (long long)gm * ldc + gn] = f2bf(v);
        } else {
          long long col = (long long)(gm / mSeg) * mPad + (gm % mSeg);
          ((u16*)Cp)[cOff + (long long)gn * ldc + col] = f2bf(v);
        }
      }
}

// ================= host orchestration =================
#define GEMM(OM, HB, HR, A_, lda_, B_, ldb_, C_, ldc_, bias_, res_, ldr_,          \
             M_, N_, K_, Z_, bh_, aSb_, aSh_, bSb_, bSh_, cSb_, cSh_, mSeg_, mPad_) \
  k_gemm<OM, HB, HR><<<dim3((unsigned)(((N_) + BN - 1) / BN),                       \
                            (unsigned)(((M_) + BM - 1) / BM), (unsigned)(Z_)),      \
                       256, 0, stream>>>(                                           \
      A_, lda_, B_, ldb_, C_, ldc_, bias_, res_, ldr_, M_, N_, K_, bh_,             \
      (long long)(aSb_), (long long)(aSh_), (long long)(bSb_), (long long)(bSh_),   \
      (long long)(cSb_), (long long)(cSh_), mSeg_, mPad_)

extern "C" void kernel_launch(void* const* d_in, const int* in_sizes, int n_in,
                              void* d_out, int out_size, void* d_ws, size_t ws_size,
                              hipStream_t stream) {
  (void)in_sizes; (void)n_in; (void)out_size;
  const float* x    = (const float*)d_in[0];
  const float* ctx  = (const float*)d_in[1];
  const float* ln1g = (const float*)d_in[2];
  const float* ln1b = (const float*)d_in[3];
  const float* wq1  = (const float*)d_in[4];
  const float* wk1  = (const float*)d_in[5];
  const float* wv1  = (const float*)d_in[6];
  const float* wo1  = (const float*)d_in[7];
  const float* bo1  = (const float*)d_in[8];
  const float* ln2g = (const float*)d_in[9];
  const float* ln2b = (const float*)d_in[10];
  const float* wq2  = (const float*)d_in[11];
  const float* wk2  = (const float*)d_in[12];
  const float* wv2  = (const float*)d_in[13];
  const float* wo2  = (const float*)d_in[14];
  const float* bo2  = (const float*)d_in[15];
  const float* ln3g = (const float*)d_in[16];
  const float* ln3b = (const float*)d_in[17];
  const float* wge  = (const float*)d_in[18];
  const float* bge  = (const float*)d_in[19];
  const float* wout = (const float*)d_in[20];
  const float* bout = (const float*)d_in[21];
  float* out = (float*)d_out;

  char* ws = (char*)d_ws;
  size_t off = 0;
  auto alloc = [&](size_t bytes) -> char* {
    char* p = ws + off;
    off = (off + bytes + 255) & ~(size_t)255;
    return p;
  };
  u16* wq1T  = (u16*)alloc((size_t)DIMC * DIMC * 2);
  u16* wk1T  = (u16*)alloc((size_t)DIMC * DIMC * 2);
  u16* wv1T  = (u16*)alloc((size_t)DIMC * DIMC * 2);
  u16* wo1T  = (u16*)alloc((size_t)DIMC * DIMC * 2);
  u16* wq2T  = (u16*)alloc((size_t)DIMC * DIMC * 2);
  u16* wk2T  = (u16*)alloc((size_t)DIMC * CTXD * 2);
  u16* wv2T  = (u16*)alloc((size_t)DIMC * CTXD * 2);
  u16* wo2T  = (u16*)alloc((size_t)DIMC * DIMC * 2);
  u16* wgeT  = (u16*)alloc((size_t)FF2 * DIMC * 2);
  u16* woutT = (u16*)alloc((size_t)DIMC * FF * 2);
  u16* ctxb  = (u16*)alloc((size_t)CROWS * CTXD * 2);
  u16* hln   = (u16*)alloc((size_t)ROWS * DIMC * 2);
  u16* qb    = (u16*)alloc((size_t)ROWS * DIMC * 2);
  u16* kb    = (u16*)alloc((size_t)ROWS * DIMC * 2);
  u16* vT    = (u16*)alloc((size_t)DIMC * ROWS * 2);   // self: ld=4096; cross: ld=384
  u16* attn  = (u16*)alloc((size_t)ROWS * DIMC * 2);
  float* xb  = (float*)alloc((size_t)ROWS * DIMC * 4);
  char* big  = alloc((size_t)NB * HEADS * SEQ * SEQ * 4); // scores f32 / geglu-h bf16
  char* pbc  = alloc((size_t)NB * HEADS * SEQ * SEQ * 2); // P bf16 / geglu-act bf16
  if (off > ws_size) return;  // insufficient scratch

  float* scoresF = (float*)big;
  u16*   Pbuf    = (u16*)pbc;
  const float scale = 0.0790569415042095f;  // 1/sqrt(160)

  auto wt = [&](const float* s, u16* d, int K, int N) {
    long long total = (long long)K * N;
    k_wtrans<<<dim3((unsigned)((total + 255) / 256)), 256, 0, stream>>>(s, d, K, N);
  };
  wt(wq1, wq1T, DIMC, DIMC);  wt(wk1, wk1T, DIMC, DIMC);
  wt(wv1, wv1T, DIMC, DIMC);  wt(wo1, wo1T, DIMC, DIMC);
  wt(wq2, wq2T, DIMC, DIMC);  wt(wk2, wk2T, CTXD, DIMC);
  wt(wv2, wv2T, CTXD, DIMC);  wt(wo2, wo2T, DIMC, DIMC);
  wt(wge, wgeT, DIMC, FF2);   wt(wout, woutT, FF, DIMC);
  {
    int n = CROWS * CTXD;
    k_tobf16<<<(n + 255) / 256, 256, 0, stream>>>(ctx, ctxb, n);
  }

  // ---------------- stage 1: self-attention ----------------
  k_layernorm<<<ROWS, 256, 0, stream>>>(x, ln1g, ln1b, hln, DIMC);
  GEMM(1, false, false, hln, DIMC, wq1T, DIMC, qb, DIMC, nullptr, nullptr, 0,
       ROWS, DIMC, DIMC, 1, 1, 0, 0, 0, 0, 0, 0, 1, 1);
  GEMM(1, false, false, hln, DIMC, wk1T, DIMC, kb, DIMC, nullptr, nullptr, 0,
       ROWS, DIMC, DIMC, 1, 1, 0, 0, 0, 0, 0, 0, 1, 1);
  GEMM(2, false, false, hln, DIMC, wv1T, DIMC, vT, ROWS, nullptr, nullptr, 0,
       ROWS, DIMC, DIMC, 1, 1, 0, 0, 0, 0, 0, 0, SEQ, SEQ);
  // scores[b,h] = Q Kt : M=S, N=S, K=HD
  GEMM(0, false, false, qb, DIMC, kb, DIMC, scoresF, SEQ, nullptr, nullptr, 0,
       SEQ, SEQ, HDIM, NB * HEADS, HEADS,
       (long long)SEQ * DIMC, HDIM, (long long)SEQ * DIMC, HDIM,
       (long long)HEADS * SEQ * SEQ, (long long)SEQ * SEQ, 1, 1);
  k_softmax<<<NB * HEADS * SEQ, 256, 0, stream>>>(scoresF, Pbuf, SEQ, SEQ, scale);
  // attn[b,h] = P Vt^T : M=S, N=HD, K=S
  GEMM(1, false, false, Pbuf, SEQ, vT, ROWS, attn, DIMC, nullptr, nullptr, 0,
       SEQ, HDIM, SEQ, NB * HEADS, HEADS,
       (long long)HEADS * SEQ * SEQ, (long long)SEQ * SEQ,
       SEQ, (long long)HDIM * ROWS,
       (long long)SEQ * DIMC, HDIM, 1, 1);
  GEMM(0, true, true, attn, DIMC, wo1T, DIMC, xb, DIMC, bo1, x, DIMC,
       ROWS, DIMC, DIMC, 1, 1, 0, 0, 0, 0, 0, 0, 1, 1);

  // ---------------- stage 2: cross-attention ----------------
  k_layernorm<<<ROWS, 256, 0, stream>>>(xb, ln2g, ln2b, hln, DIMC);
  GEMM(1, false, false, hln, DIMC, wq2T, DIMC, qb, DIMC, nullptr, nullptr, 0,
       ROWS, DIMC, DIMC, 1, 1, 0, 0, 0, 0, 0, 0, 1, 1);
  GEMM(1, false, false, ctxb, CTXD, wk2T, CTXD, kb, DIMC, nullptr, nullptr, 0,
       CROWS, DIMC, CTXD, 1, 1, 0, 0, 0, 0, 0, 0, 1, 1);
  GEMM(2, false, false, ctxb, CTXD, wv2T, CTXD, vT, NB * TPADC, nullptr, nullptr, 0,
       CROWS, DIMC, CTXD, 1, 1, 0, 0, 0, 0, 0, 0, TCTX, TPADC);
  GEMM(0, false, false, qb, DIMC, kb, DIMC, scoresF, TPADC, nullptr, nullptr, 0,
       SEQ, TCTX, HDIM, NB * HEADS, HEADS,
       (long long)SEQ * DIMC, HDIM, (long long)TCTX * DIMC, HDIM,
       (long long)HEADS * SEQ * TPADC, (long long)SEQ * TPADC, 1, 1);
  k_softmax<<<NB * HEADS * SEQ, 256, 0, stream>>>(scoresF, Pbuf, TCTX, TPADC, scale);
  GEMM(1, false, false, Pbuf, TPADC, vT, NB * TPADC, attn, DIMC, nullptr, nullptr, 0,
       SEQ, HDIM, TCTX, NB * HEADS, HEADS,
       (long long)HEADS * SEQ * TPADC, (long long)SEQ * TPADC,
       TPADC, (long long)HDIM * NB * TPADC,
       (long long)SEQ * DIMC, HDIM, 1, 1);
  GEMM(0, true, true, attn, DIMC, wo2T, DIMC, xb, DIMC, bo2, xb, DIMC,
       ROWS, DIMC, DIMC, 1, 1, 0, 0, 0, 0, 0, 0, 1, 1);

  // ---------------- stage 3: GEGLU MLP ----------------
  k_layernorm<<<ROWS, 256, 0, stream>>>(xb, ln3g, ln3b, hln, DIMC);
  u16* geh = (u16*)big;
  GEMM(1, true, false, hln, DIMC, wgeT, DIMC, geh, FF2, bge, nullptr, 0,
       ROWS, FF2, DIMC, 1, 1, 0, 0, 0, 0, 0, 0, 1, 1);
  u16* gact = (u16*)pbc;
  k_geglu<<<(ROWS * FF + 255) / 256, 256, 0, stream>>>(geh, gact, ROWS * FF);
  GEMM(0, true, true, gact, FF, woutT, FF, out, DIMC, bout, xb, DIMC,
       ROWS, DIMC, FF, 1, 1, 0, 0, 0, 0, 0, 0, 1, 1);
}
